// GATPredictor_49194555408820
// MI455X (gfx1250) — matrix operand order repaired and checked by reading, not compiled, not run on hardware
//
#include <hip/hip_runtime.h>
#include <math.h>

typedef __attribute__((ext_vector_type(16))) __bf16 v16bf;
typedef __attribute__((ext_vector_type(8)))  __bf16 v8bf;
typedef __attribute__((ext_vector_type(8)))  float  v8f;

#define INPUT_DIM 256
#define HIDDEN    64

__device__ __forceinline__ __bf16 f2bf(float f) { return (__bf16)f; }

// ---------------------------------------------------------------------------
// Kernel 1: h = relu(relu(x@W1+b1)@W2+b2), all in bf16 WMMA with f32 accum.
// 256 threads = 8 waves; each wave owns one 16-node tile (16x64 output).
// ---------------------------------------------------------------------------
__global__ __launch_bounds__(256)
void mlp_wmma_kernel(const float* __restrict__ x,
                     const float* __restrict__ W1, const float* __restrict__ b1,
                     const float* __restrict__ W2, const float* __restrict__ b2,
                     float* __restrict__ h, int nTiles)
{
    // W1/W2 staged in per-lane B-fragment layout:
    //   index = ((kstep*4 + ntile)*32 + lane)*16 + (2*j + hi)
    //   holds W[k][col] with col = ntile*16 + (lane&15),
    //                        k  = kstep*32 + (lane>>4)*8 + (p&7) + (p>>3)*16, p = 2*j + hi
    __shared__ __align__(32) __bf16 Wb1[8 * 4 * 32 * 16];   // 32 KB
    __shared__ __align__(32) __bf16 Wb2[2 * 4 * 32 * 16];   //  8 KB
    __shared__ __align__(32) __bf16 h1s[8][16 * HIDDEN];    // 16 KB (per-wave 16x64)

    const int tid = threadIdx.x;

    for (int i = tid; i < 8 * 4 * 32 * 16; i += 256) {
        int p     = i & 15;
        int lane_ = (i >> 4) & 31;
        int frag  = i >> 9;
        int kstep = frag >> 2, ntile = frag & 3;
        int col   = ntile * 16 + (lane_ & 15);
        int k     = kstep * 32 + (lane_ >> 4) * 8 + (p & 7) + (p >> 3) * 16;
        Wb1[i] = f2bf(W1[k * HIDDEN + col]);
    }
    for (int i = tid; i < 2 * 4 * 32 * 16; i += 256) {
        int p     = i & 15;
        int lane_ = (i >> 4) & 31;
        int frag  = i >> 9;
        int kstep = frag >> 2, ntile = frag & 3;
        int col   = ntile * 16 + (lane_ & 15);
        int k     = kstep * 32 + (lane_ >> 4) * 8 + (p & 7) + (p >> 3) * 16;
        Wb2[i] = f2bf(W2[k * HIDDEN + col]);
    }
    __syncthreads();

    const int wave = tid >> 5;
    const int lane = tid & 31;
    const int tile = blockIdx.x * 8 + wave;
    if (tile >= nTiles) return;              // wave-uniform: EXEC stays all-1s

    const int row = lane & 15;               // M row this lane feeds in A
    const int sel = lane >> 4;               // K-half selector
    const int nodeBase = tile * 16;
    const float* xrow = x + (size_t)(nodeBase + row) * INPUT_DIM;

    // ---------------- Layer 1: 16x256 @ 256x64 ----------------
    v8f c[4] = {v8f{}, v8f{}, v8f{}, v8f{}};

    #pragma unroll
    for (int ks = 0; ks < 8; ++ks) {
        const int k0 = ks * 32;
        // A fragment (16x32 bf16): lane reads two contiguous 32B float chunks
        const float4* p0 = (const float4*)(xrow + k0 + sel * 8);
        const float4* p1 = (const float4*)(xrow + k0 + 16 + sel * 8);
        float4 f0 = p0[0], f1 = p0[1], f2 = p1[0], f3 = p1[1];
        v16bf a;
        a[0]  = f2bf(f0.x); a[1]  = f2bf(f0.y); a[2]  = f2bf(f0.z); a[3]  = f2bf(f0.w);
        a[4]  = f2bf(f1.x); a[5]  = f2bf(f1.y); a[6]  = f2bf(f1.z); a[7]  = f2bf(f1.w);
        a[8]  = f2bf(f2.x); a[9]  = f2bf(f2.y); a[10] = f2bf(f2.z); a[11] = f2bf(f2.w);
        a[12] = f2bf(f3.x); a[13] = f2bf(f3.y); a[14] = f2bf(f3.z); a[15] = f2bf(f3.w);

        #pragma unroll
        for (int nt = 0; nt < 4; ++nt) {
            const v16bf b = *(const v16bf*)&Wb1[((ks * 4 + nt) * 32 + lane) * 16];
            c[nt] = __builtin_amdgcn_wmma_f32_16x16x32_bf16(
                        false, a, false, b, (short)0, c[nt], false, false);
        }
    }

    // bias + relu, reshape C-layout -> A-layout via per-wave LDS tile (bf16)
    __bf16* my = &h1s[wave][0];
    #pragma unroll
    for (int nt = 0; nt < 4; ++nt) {
        const int col = nt * 16 + (lane & 15);
        const float bb = b1[col];
        #pragma unroll
        for (int r = 0; r < 8; ++r) {
            float v = fmaxf(c[nt][r] + bb, 0.0f);
            my[(r + sel * 8) * HIDDEN + col] = f2bf(v);
        }
    }
    // per-wave DS RAW: keep order and drain DS before re-reading
    __builtin_amdgcn_sched_barrier(0);
    asm volatile("s_wait_dscnt 0" ::: "memory");

    // ---------------- Layer 2: 16x64 @ 64x64 ----------------
    v8f c2[4] = {v8f{}, v8f{}, v8f{}, v8f{}};
    #pragma unroll
    for (int ks = 0; ks < 2; ++ks) {
        const int k0 = ks * 32;
        const v8bf lo = *(const v8bf*)&my[row * HIDDEN + k0 + sel * 8];
        const v8bf hi = *(const v8bf*)&my[row * HIDDEN + k0 + 16 + sel * 8];
        v16bf a;
        #pragma unroll
        for (int i = 0; i < 8; ++i) { a[i] = lo[i]; a[8 + i] = hi[i]; }

        #pragma unroll
        for (int nt = 0; nt < 4; ++nt) {
            const v16bf b = *(const v16bf*)&Wb2[((ks * 4 + nt) * 32 + lane) * 16];
            c2[nt] = __builtin_amdgcn_wmma_f32_16x16x32_bf16(
                         false, a, false, b, (short)0, c2[nt], false, false);
        }
    }

    // bias + relu + store h tile (fp32)
    #pragma unroll
    for (int nt = 0; nt < 4; ++nt) {
        const int col = nt * 16 + (lane & 15);
        const float bb = b2[col];
        #pragma unroll
        for (int r = 0; r < 8; ++r) {
            float v = fmaxf(c2[nt][r] + bb, 0.0f);
            h[(size_t)(nodeBase + r + sel * 8) * HIDDEN + col] = v;
        }
    }
}

// ---------------------------------------------------------------------------
// Kernel 2: per-edge score = sigmoid(dot(h[s], h[t]) * w). 16 lanes per edge,
// float4 gathers (h is L2-resident: 25.6 MB << 192 MB), shuffle reduce.
// ---------------------------------------------------------------------------
__global__ __launch_bounds__(256)
void edge_score_kernel(const int* __restrict__ src, const int* __restrict__ dst,
                       const float* __restrict__ w, const float* __restrict__ h,
                       float* __restrict__ out, int nEdges)
{
    const int t = blockIdx.x * 256 + threadIdx.x;
    const int e = t >> 4;
    const int l = t & 15;
    if (e >= nEdges) return;

    const int s = src[e];
    const int d = dst[e];
    const float4 a = ((const float4*)(h + (size_t)s * HIDDEN))[l];
    const float4 b = ((const float4*)(h + (size_t)d * HIDDEN))[l];
    float p = a.x * b.x + a.y * b.y + a.z * b.z + a.w * b.w;

    p += __shfl_xor(p, 8, 32);
    p += __shfl_xor(p, 4, 32);
    p += __shfl_xor(p, 2, 32);
    p += __shfl_xor(p, 1, 32);

    if (l == 0) {
        const float z = p * w[e];
        out[e] = 1.0f / (1.0f + __expf(-z));
    }
}

extern "C" void kernel_launch(void* const* d_in, const int* in_sizes, int n_in,
                              void* d_out, int out_size, void* d_ws, size_t ws_size,
                              hipStream_t stream)
{
    const float* x  = (const float*)d_in[0];
    const int*   ei = (const int*)  d_in[1];   // [2, E] int32
    const float* wt = (const float*)d_in[2];
    const float* W1 = (const float*)d_in[3];
    const float* b1 = (const float*)d_in[4];
    const float* W2 = (const float*)d_in[5];
    const float* b2 = (const float*)d_in[6];
    float* out  = (float*)d_out;
    float* hbuf = (float*)d_ws;                // 100000*64 fp32 = 25.6 MB

    const int nNodes = in_sizes[0] / INPUT_DIM;
    const int nEdges = in_sizes[2];
    const int nTiles = nNodes / 16;            // 6250 (exact)

    const int mlpBlocks = (nTiles + 7) / 8;
    mlp_wmma_kernel<<<mlpBlocks, 256, 0, stream>>>(x, W1, b1, W2, b2, hbuf, nTiles);

    const long long eThreads = (long long)nEdges * 16;
    const int edgeBlocks = (int)((eThreads + 255) / 256);
    edge_score_kernel<<<edgeBlocks, 256, 0, stream>>>(ei, ei + nEdges, wt, hbuf, out, nEdges);
}
